// SelectiveSSM_51007031607475
// MI455X (gfx1250) — matrix operand order repaired and checked by reading, not compiled, not run on hardware
//
#include <hip/hip_runtime.h>

#define BSZ    8
#define SEQ    2048
#define DMODEL 768
#define DSTATE 64
#define DTRANK 128
#define DCONV  4

typedef __attribute__((ext_vector_type(16))) __bf16 v16bf;
typedef __attribute__((ext_vector_type(8)))  float  v8f;
typedef __attribute__((ext_vector_type(4)))  float  v4f;
typedef __attribute__((ext_vector_type(4)))  unsigned int v4u;
typedef __attribute__((ext_vector_type(8)))  int    v8i;
typedef __attribute__((ext_vector_type(4)))  int    v4i;

union BF16Frag { v16bf v; unsigned short u[16]; };

#if __has_builtin(__builtin_amdgcn_tensor_load_to_lds)
#define HAVE_TDM 1
#else
#define HAVE_TDM 0
#endif

__device__ __forceinline__ unsigned short f32_to_bf16(float x) {
  unsigned int b = __float_as_uint(x);
  b += 0x7FFFu + ((b >> 16) & 1u);            // round to nearest even
  return (unsigned short)(b >> 16);
}
__device__ __forceinline__ float bf16_to_f32(unsigned short h) {
  return __uint_as_float(((unsigned int)h) << 16);
}

// ---------------------------------------------------------------------------
// Pack W (K x N row-major f32) into WMMA B-fragment order, hi/lo bf16 split.
// Fragment (nt, kt): lane 0-15 holds K=kt*32+0..15 of col n=nt*16+lane,
//                    lane 16-31 holds K=kt*32+16..31 of col n=nt*16+lane-16.
// Per lane: 16 hi bf16 then 16 lo bf16 (64 bytes contiguous).
// ---------------------------------------------------------------------------
__global__ void pack_weights_kernel(const float* __restrict__ W,
                                    unsigned short* __restrict__ P,
                                    int K, int N) {
  int t = blockIdx.x * 256 + threadIdx.x;
  int KT = K >> 5;
  int total = KT * (N >> 4) * 32;
  if (t >= total) return;
  int lane = t & 31;
  int frag = t >> 5;
  int kt = frag % KT;
  int nt = frag / KT;
  int n = (nt << 4) + (lane & 15);
  int kbase = (kt << 5) + ((lane & 16) ? 16 : 0);
  unsigned short* dst = P + (size_t)t * 32;   // t == (nt*KT+kt)*32 + lane
#pragma unroll
  for (int i = 0; i < 16; ++i) {
    float w = W[(size_t)(kbase + i) * N + n];
    unsigned short h = f32_to_bf16(w);
    dst[i] = h;
    dst[16 + i] = f32_to_bf16(w - bf16_to_f32(h));
  }
}

// ---------------------------------------------------------------------------
// Causal depthwise conv1d (left pad K-1) + bias + SiLU. One thread per elem.
// ---------------------------------------------------------------------------
__global__ __launch_bounds__(256) void conv_silu_kernel(
    const float* __restrict__ x, const float* __restrict__ cw,
    const float* __restrict__ cb, float* __restrict__ xc) {
  size_t idx = (size_t)blockIdx.x * 256 + threadIdx.x;
  int d = (int)(idx % DMODEL);
  size_t t = idx / DMODEL;
  int l = (int)(t % SEQ);
  int b = (int)(t / SEQ);
  float acc = cb[d];
#pragma unroll
  for (int k = 0; k < DCONV; ++k) {
    int ls = l - (DCONV - 1) + k;
    if (ls >= 0)
      acc = fmaf(x[((size_t)b * SEQ + ls) * DMODEL + d], cw[d * DCONV + k], acc);
  }
  xc[idx] = acc / (1.0f + __expf(-acc));      // SiLU
}

#if HAVE_TDM
// ---------------------------------------------------------------------------
// TDM: async-copy one k-step of packed B weights (4 n-subtiles) into LDS.
// 2-D tile: 4 rows x 2048 bytes, row stride KT*2048 bytes. data_size = 8 B.
// Descriptor per CDNA5 ISA. Tracked by TENSORcnt.
// ---------------------------------------------------------------------------
__device__ __forceinline__ void tdm_load_b_tile(const unsigned short* gsrc,
                                                unsigned int lds_byte_off,
                                                int KT) {
  unsigned long long ga = (unsigned long long)(size_t)gsrc;
  v4u g0;
  g0[0] = 1u;                                          // count=1 (valid user D#)
  g0[1] = lds_byte_off;                                // lds_addr
  g0[2] = (unsigned int)(ga & 0xFFFFFFFFull);          // global_addr[31:0]
  g0[3] = (unsigned int)((ga >> 32) & 0x1FFFFFFull)    // global_addr[56:32]
          | (2u << 30);                                // type = 2 ("image")
  const unsigned int td0   = (unsigned int)KT * 256u;  // tensor_dim0 (8B units)
  const unsigned int td1   = 16u;                      // tensor_dim1 (>= tile1)
  const unsigned int tile0 = 256u;                     // 2048 B row, 8B units
  const unsigned int tile1 = 4u;                       // 4 n-subtiles
  const unsigned long long s0 = (unsigned long long)KT * 256ull; // dim0 stride
  v8i g1;
  g1[0] = (int)(3u << 16);                             // mask=0, data_size=8B
  g1[1] = (int)((td0 & 0xFFFFu) << 16);                // tensor_dim0[15:0]
  g1[2] = (int)(((td0 >> 16) & 0xFFFFu) | ((td1 & 0xFFFFu) << 16));
  g1[3] = (int)(((td1 >> 16) & 0xFFFFu) | (tile0 << 16));  // tile_dim0
  g1[4] = (int)(tile1 & 0xFFFFu);                      // tile_dim1, tile_dim2=0
  g1[5] = (int)(s0 & 0xFFFFFFFFull);                   // dim0_stride[31:0]
  g1[6] = (int)((s0 >> 32) & 0xFFFFull);               // dim0_stride[47:32]
  g1[7] = 0;                                           // dim1_stride unused
  v4i z = {0, 0, 0, 0};                                // groups 2/3: <=2-D tile
#if __clang_major__ >= 23
  v8i z8 = {0, 0, 0, 0, 0, 0, 0, 0};
  __builtin_amdgcn_tensor_load_to_lds(g0, g1, z, z, z8, 0);
#else
  __builtin_amdgcn_tensor_load_to_lds(g0, g1, z, z, 0);
#endif
}
#endif  // HAVE_TDM

// ---------------------------------------------------------------------------
// GEMM  C(M x N) = A(M x K, f32 row-major, leading dim lda) * Wpacked
// 3x bf16-split WMMA per k-step: hi*hi + hi*lo + lo*hi  (~fp32 accuracy).
// Block: 256 threads = 8 waves; wave tile 16(M) x 64(N); block 128M x 64N.
// B weight tiles staged through LDS by the Tensor Data Mover, double-buffered
// (all 8 waves share the same B fragments -> 8x global-traffic reduction).
// act: 0 = none, 1 = softplus (with bias).
// ---------------------------------------------------------------------------
__global__ __launch_bounds__(256) void gemm_bf16x3_kernel(
    const float* __restrict__ A, int lda,
    const unsigned short* __restrict__ Wp,
    const float* __restrict__ bias,
    float* __restrict__ C, int ldc,
    int K, int act) {
  const int lane = threadIdx.x & 31;
  const int wave = threadIdx.x >> 5;
  const int m0 = blockIdx.y * 128 + wave * 16;
  const int n0 = blockIdx.x * 64;
  const int KT = K >> 5;
  const int nt0 = n0 >> 4;
  const int m = m0 + (lane & 15);
  const int aoff = (lane & 16) ? 16 : 0;      // A frag: K in {0..15} or {16..31}

  v8f acc[4] = {};

#if HAVE_TDM
  __shared__ __align__(32) unsigned short ldsB[2][4096];   // 2 x 8 KB
  // prologue: stage k-step 0 into buffer 0
  if (threadIdx.x < 32) {
    tdm_load_b_tile(Wp + (size_t)(nt0 * KT) * 1024,
                    (unsigned int)(size_t)&ldsB[0][0], KT);
    __builtin_amdgcn_s_wait_tensorcnt((short)0);
  }
  __syncthreads();
#endif

  for (int kt = 0; kt < KT; ++kt) {
#if HAVE_TDM
    const int bufi = kt & 1;
    if (kt + 1 < KT && threadIdx.x < 32)      // async-stage next k-step
      tdm_load_b_tile(Wp + (size_t)(nt0 * KT + (kt + 1)) * 1024,
                      (unsigned int)(size_t)&ldsB[bufi ^ 1][0], KT);
#endif
    // ---- A fragment (on-the-fly hi/lo split) ----
    const float* ap = A + (size_t)m * lda + kt * 32 + aoff;
    v4f a0 = *(const v4f*)(ap);
    v4f a1 = *(const v4f*)(ap + 4);
    v4f a2 = *(const v4f*)(ap + 8);
    v4f a3 = *(const v4f*)(ap + 12);
    float af[16] = {a0.x, a0.y, a0.z, a0.w, a1.x, a1.y, a1.z, a1.w,
                    a2.x, a2.y, a2.z, a2.w, a3.x, a3.y, a3.z, a3.w};
    BF16Frag ahi, alo;
#pragma unroll
    for (int i = 0; i < 16; ++i) {
      unsigned short h = f32_to_bf16(af[i]);
      ahi.u[i] = h;
      alo.u[i] = f32_to_bf16(af[i] - bf16_to_f32(h));
    }
    // ---- 4 N-subtiles ----
#pragma unroll
    for (int j = 0; j < 4; ++j) {
#if HAVE_TDM
      const BF16Frag* bp =
          (const BF16Frag*)(&ldsB[bufi][j * 1024 + lane * 32]);
      BF16Frag bhi = bp[0];
      BF16Frag blo = bp[1];
#else
      const unsigned short* bg =
          Wp + ((size_t)((nt0 + j) * KT + kt) * 32 + lane) * 32;
      BF16Frag bhi = *(const BF16Frag*)bg;
      BF16Frag blo = *(const BF16Frag*)(bg + 16);
#endif
      acc[j] = __builtin_amdgcn_wmma_f32_16x16x32_bf16(
          false, ahi.v, false, bhi.v, (short)0, acc[j], false, false);
      acc[j] = __builtin_amdgcn_wmma_f32_16x16x32_bf16(
          false, ahi.v, false, blo.v, (short)0, acc[j], false, false);
      acc[j] = __builtin_amdgcn_wmma_f32_16x16x32_bf16(
          false, alo.v, false, bhi.v, (short)0, acc[j], false, false);
    }
#if HAVE_TDM
    if (kt + 1 < KT) {
      if (threadIdx.x < 32) __builtin_amdgcn_s_wait_tensorcnt((short)0);
      __syncthreads();                        // release buffer bufi^1
    }
#endif
  }

  // ---- epilogue: C/D layout -> VGPR r: lanes0-15 M=r, lanes16-31 M=r+8 ----
  const int rowoff = (lane & 16) ? 8 : 0;
#pragma unroll
  for (int j = 0; j < 4; ++j) {
    int col = n0 + j * 16 + (lane & 15);
    float bv = bias ? bias[col] : 0.0f;
#pragma unroll
    for (int r = 0; r < 8; ++r) {
      int row = m0 + r + rowoff;
      float v = acc[j][r] + bv;
      if (act == 1) v = (v > 20.0f) ? v : log1pf(__expf(v));   // softplus
      C[(size_t)row * ldc + col] = v;
    }
  }
}

// ---------------------------------------------------------------------------
// Selective scan: one wave per (b, d) recurrence; lane holds states n=2*lane,
// 2*lane+1.  y[b,l,d] = sum_n state * C via shfl_xor butterfly reduction.
// ---------------------------------------------------------------------------
__global__ __launch_bounds__(128) void scan_kernel(
    const float* __restrict__ xc, const float* __restrict__ xdbl,
    const float* __restrict__ dt, const float* __restrict__ A_log,
    float* __restrict__ y) {
  const int lane = threadIdx.x & 31;
  const int gw = blockIdx.x * 4 + (threadIdx.x >> 5);   // 0 .. B*D-1
  const int b = gw / DMODEL;
  const int d = gw - b * DMODEL;
  const int n0 = lane * 2;

  const float A0 = -__expf(A_log[d * DSTATE + n0]);
  const float A1 = -__expf(A_log[d * DSTATE + n0 + 1]);
  float s0 = 0.0f, s1 = 0.0f;

  const float* xdb = xdbl + (size_t)b * SEQ * 256;
  const float* dtb = dt + (size_t)b * SEQ * DMODEL + d;
  const float* ub  = xc + (size_t)b * SEQ * DMODEL + d;
  float*       yb  = y  + (size_t)b * SEQ * DMODEL + d;

  for (int l = 0; l < SEQ; ++l) {
    const float* row = xdb + (size_t)l * 256;
    float dt_v = dtb[(size_t)l * DMODEL];
    float u_v  = ub[(size_t)l * DMODEL];
    float2 Bp = *(const float2*)(row + 128 + 2 * lane);
    float2 Cp = *(const float2*)(row + 192 + 2 * lane);

    // speculative prefetch a few timesteps ahead (global_prefetch_b8)
    __builtin_prefetch(row + 8 * 256, 0, 0);
    __builtin_prefetch(dtb + (size_t)(l + 8) * DMODEL, 0, 0);
    __builtin_prefetch(ub + (size_t)(l + 8) * DMODEL, 0, 0);

    float dA0 = __expf(dt_v * A0);
    float dA1 = __expf(dt_v * A1);
    float du = dt_v * u_v;
    s0 = fmaf(dA0, s0, du * Bp.x);
    s1 = fmaf(dA1, s1, du * Bp.y);
    float yp = fmaf(s0, Cp.x, s1 * Cp.y);
#pragma unroll
    for (int off = 16; off > 0; off >>= 1) yp += __shfl_xor(yp, off, 32);
    if (lane == 0) yb[(size_t)l * DMODEL] = yp;
  }
}

// ---------------------------------------------------------------------------
extern "C" void kernel_launch(void* const* d_in, const int* in_sizes, int n_in,
                              void* d_out, int out_size, void* d_ws, size_t ws_size,
                              hipStream_t stream) {
  const float* x         = (const float*)d_in[0];
  const float* A_log     = (const float*)d_in[1];
  const float* x_proj_w  = (const float*)d_in[2];
  const float* dt_proj_w = (const float*)d_in[3];
  const float* dt_proj_b = (const float*)d_in[4];
  const float* conv_w    = (const float*)d_in[5];
  const float* conv_b    = (const float*)d_in[6];
  const float* out_w     = (const float*)d_in[7];
  float* out = (float*)d_out;

  const size_t BLD = (size_t)BSZ * SEQ * DMODEL;   // 12,582,912
  const size_t BL  = (size_t)BSZ * SEQ;            // 16,384

  // workspace layout (~171 MB)
  float* xc   = (float*)d_ws;
  float* xdbl = xc + BLD;                          // BL x 256
  float* dtb  = xdbl + BL * 256;                   // BL x D
  float* yb   = dtb + BLD;                         // BL x D
  unsigned short* w1p = (unsigned short*)(yb + BLD);
  unsigned short* w2p = w1p + (size_t)DMODEL * 256 * 2;     // hi+lo
  unsigned short* w3p = w2p + (size_t)DTRANK * DMODEL * 2;

  // pack weights into WMMA B-fragment order (hi/lo bf16)
  int tot1 = (DMODEL / 32) * (256 / 16) * 32;
  pack_weights_kernel<<<(tot1 + 255) / 256, 256, 0, stream>>>(x_proj_w, w1p, DMODEL, 256);
  int tot2 = (DTRANK / 32) * (DMODEL / 16) * 32;
  pack_weights_kernel<<<(tot2 + 255) / 256, 256, 0, stream>>>(dt_proj_w, w2p, DTRANK, DMODEL);
  int tot3 = (DMODEL / 32) * (DMODEL / 16) * 32;
  pack_weights_kernel<<<(tot3 + 255) / 256, 256, 0, stream>>>(out_w, w3p, DMODEL, DMODEL);

  // 1) causal depthwise conv + SiLU
  conv_silu_kernel<<<(unsigned)(BLD / 256), 256, 0, stream>>>(x, conv_w, conv_b, xc);

  // 2) GEMM1: xc (BL x 768) @ W1 (768 x 256) -> x_dbl
  gemm_bf16x3_kernel<<<dim3(256 / 64, (unsigned)(BL / 128)), 256, 0, stream>>>(
      xc, DMODEL, w1p, nullptr, xdbl, 256, DMODEL, 0);

  // 3) GEMM2: x_dbl[:, :128] @ W2 (128 x 768) + bias, softplus -> dt
  gemm_bf16x3_kernel<<<dim3(DMODEL / 64, (unsigned)(BL / 128)), 256, 0, stream>>>(
      xdbl, 256, w2p, dt_proj_b, dtb, DMODEL, DTRANK, 1);

  // 4) selective scan -> y
  scan_kernel<<<(BSZ * DMODEL) / 4, 128, 0, stream>>>(xc, xdbl, dtb, A_log, yb);

  // 5) GEMM3: y (BL x 768) @ out_w (768 x 768) -> out
  gemm_bf16x3_kernel<<<dim3(DMODEL / 64, (unsigned)(BL / 128)), 256, 0, stream>>>(
      yb, DMODEL, w3p, nullptr, out, DMODEL, DMODEL, 0);
}
